// PointNetSetAbstraction_26362509263294
// MI455X (gfx1250) — matrix operand order repaired and checked by reading, not compiled, not run on hardware
//
#include <hip/hip_runtime.h>
#include <hip/hip_bf16.h>

// ---------------------------------------------------------------------------
// PointNet++ Set Abstraction for MI455X (gfx1250, wave32, WMMA bf16)
//   B=16, N=4096, C=64, NPOINT=1024, NSAMPLE=32, mlp = [64,64,128]
// ---------------------------------------------------------------------------

#define BATCH   16
#define NPTS    4096
#define CIN_PTS 64
#define NPOINT  1024
#define NSAMPLE 32
#define MROWS   (BATCH * NPOINT * NSAMPLE)   // 524288
#define RAD2    0.04f
#define BN_EPS  1e-5f

typedef __attribute__((ext_vector_type(16))) __bf16 v16bf;
typedef __attribute__((ext_vector_type(8)))  __bf16 bf16x8;
typedef __attribute__((ext_vector_type(8)))  float  v8f;

// ---------------------------------------------------------------------------
// 1) Farthest point sampling: one block per batch, xyz + dist cached in LDS.
//    Records farthest BEFORE updating (matches jax.lax.scan output), ties go
//    to the smallest index (jnp.argmax first-occurrence semantics).
// ---------------------------------------------------------------------------
__global__ __launch_bounds__(256) void fps_kernel(const float* __restrict__ xyz,
                                                  int* __restrict__ fps_idx,
                                                  float* __restrict__ new_xyz)
{
    __shared__ float sx[NPTS], sy[NPTS], sz[NPTS];
    __shared__ float dist[NPTS];
    __shared__ float rv[256];
    __shared__ int   ri[256];
    __shared__ int   s_far;
    __shared__ float s_cx, s_cy, s_cz;

    const int b = blockIdx.x;
    const int tid = threadIdx.x;
    const float* xb = xyz + (size_t)b * NPTS * 3;

    for (int i = tid; i < NPTS; i += 256) {
        sx[i] = xb[i * 3 + 0];
        sy[i] = xb[i * 3 + 1];
        sz[i] = xb[i * 3 + 2];
        dist[i] = 1e10f;
    }
    if (tid == 0) s_far = 0;
    __syncthreads();

    for (int t = 0; t < NPOINT; ++t) {
        if (tid == 0) {
            int far = s_far;
            fps_idx[b * NPOINT + t] = far;
            float x = sx[far], y = sy[far], z = sz[far];
            float* o = new_xyz + ((size_t)b * NPOINT + t) * 3;
            o[0] = x; o[1] = y; o[2] = z;
            s_cx = x; s_cy = y; s_cz = z;
        }
        __syncthreads();
        const float cx = s_cx, cy = s_cy, cz = s_cz;

        float lmax = -1.0f;
        int   li   = 0;
        for (int i = tid; i < NPTS; i += 256) {
            float dx = sx[i] - cx, dy = sy[i] - cy, dz = sz[i] - cz;
            float d  = dx * dx + dy * dy + dz * dz;
            float nd = fminf(dist[i], d);
            dist[i]  = nd;
            if (nd > lmax) { lmax = nd; li = i; }
        }
        rv[tid] = lmax; ri[tid] = li;
        __syncthreads();
        for (int off = 128; off > 0; off >>= 1) {
            if (tid < off) {
                float ov = rv[tid + off]; int oi = ri[tid + off];
                if (ov > rv[tid] || (ov == rv[tid] && oi < ri[tid])) {
                    rv[tid] = ov; ri[tid] = oi;
                }
            }
            __syncthreads();
        }
        if (tid == 0) s_far = ri[0];
        __syncthreads();
    }
}

// ---------------------------------------------------------------------------
// 2) Ball query: equivalent to "sort indices, take first NSAMPLE in ball,
//    pad with the first in-ball index". One thread per query point; the
//    batch's xyz is staged in LDS.
// ---------------------------------------------------------------------------
__global__ __launch_bounds__(256) void qbp_kernel(const float* __restrict__ xyz,
                                                  const float* __restrict__ new_xyz,
                                                  int* __restrict__ idx)
{
    __shared__ float sx[NPTS], sy[NPTS], sz[NPTS];
    const int b = blockIdx.x >> 2;
    const int squad = blockIdx.x & 3;
    const float* xb = xyz + (size_t)b * NPTS * 3;
    for (int i = threadIdx.x; i < NPTS; i += 256) {
        sx[i] = xb[i * 3 + 0];
        sy[i] = xb[i * 3 + 1];
        sz[i] = xb[i * 3 + 2];
    }
    __syncthreads();

    const int s = squad * 256 + threadIdx.x;
    const float* q = new_xyz + ((size_t)b * NPOINT + s) * 3;
    const float qx = q[0], qy = q[1], qz = q[2];
    int* op = idx + ((size_t)b * NPOINT + s) * NSAMPLE;

    int cnt = 0, first = 0;
    for (int j = 0; j < NPTS; ++j) {
        float dx = sx[j] - qx, dy = sy[j] - qy, dz = sz[j] - qz;
        float d2 = dx * dx + dy * dy + dz * dz;
        if (d2 <= RAD2) {
            if (cnt == 0) first = j;
            op[cnt++] = j;
            if (cnt == NSAMPLE) break;
        }
    }
    for (int t = cnt; t < NSAMPLE; ++t) op[t] = first;
}

// ---------------------------------------------------------------------------
// 3) Build grouped features gp: (M, 96) bf16, channels [0..2] = normalized
//    relative xyz, [3..66] = gathered point features, [67..95] = zero pad
//    (so layer-0 K dim is a multiple of 32 for 16x16x32 WMMA).
// ---------------------------------------------------------------------------
__global__ __launch_bounds__(256) void build_gp_kernel(const float* __restrict__ xyz,
                                                       const float* __restrict__ points,
                                                       const float* __restrict__ new_xyz,
                                                       const int* __restrict__ idx,
                                                       __bf16* __restrict__ X0)
{
    const size_t m = (size_t)blockIdx.x * 256 + threadIdx.x;
    if (m >= (size_t)MROWS) return;
    const size_t bs = m >> 5;          // (b * NPOINT + s)
    const int b = (int)(bs >> 10);
    const int pid = idx[m];

    const float* nx = new_xyz + bs * 3;
    const float* px = xyz + ((size_t)b * NPTS + pid) * 3;
    __bf16* row = X0 + m * 96;
    row[0] = (__bf16)((px[0] - nx[0]) * 5.0f);   // /RADIUS
    row[1] = (__bf16)((px[1] - nx[1]) * 5.0f);
    row[2] = (__bf16)((px[2] - nx[2]) * 5.0f);

    const float* pp = points + ((size_t)b * NPTS + pid) * CIN_PTS;
    #pragma unroll 8
    for (int c = 0; c < CIN_PTS; ++c) row[3 + c] = (__bf16)pp[c];
    #pragma unroll
    for (int c = 67; c < 96; ++c) row[c] = (__bf16)0.0f;
}

// ---------------------------------------------------------------------------
// 4) Pad/convert weights to bf16 (layer 0 padded 67 -> 96 with zeros).
// ---------------------------------------------------------------------------
__global__ void prep_weights_kernel(const float* __restrict__ w0,
                                    const float* __restrict__ w1,
                                    const float* __restrict__ w2,
                                    __bf16* __restrict__ W0,
                                    __bf16* __restrict__ W1,
                                    __bf16* __restrict__ W2)
{
    const int t = blockIdx.x * 256 + threadIdx.x;
    if (t < 64 * 96) {
        int o = t / 96, c = t % 96;
        W0[t] = (__bf16)(c < 67 ? w0[o * 67 + c] : 0.0f);
    }
    if (t < 64 * 64)  W1[t] = (__bf16)w1[t];
    if (t < 128 * 64) W2[t] = (__bf16)w2[t];
}

// ---------------------------------------------------------------------------
// 5) GEMM + fused BN statistics.  h[m,o] = sum_c X[m,c] * W[o,c] + bias[o]
//    Each wave computes a 16-row stripe over all NOUT columns with
//    v_wmma_f32_16x16x32_bf16; bias is pre-loaded into the C fragment and
//    per-channel sum / sum-of-squares are reduced through LDS then one
//    global atomic per channel per block.
// ---------------------------------------------------------------------------
template <int KPAD, int NOUT>
__global__ __launch_bounds__(256) void gemm_bn_stats_kernel(const __bf16* __restrict__ X,
                                                            const __bf16* __restrict__ W,
                                                            const float* __restrict__ bias,
                                                            __bf16* __restrict__ H,
                                                            float* __restrict__ stats)
{
    __shared__ float s_sum[NOUT];
    __shared__ float s_sq[NOUT];
    const int tid = threadIdx.x;
    if (tid < NOUT) { s_sum[tid] = 0.0f; s_sq[tid] = 0.0f; }
    __syncthreads();

    const int wave = tid >> 5;
    const int lane = tid & 31;
    const int lrow = lane & 15;
    const int half = lane >> 4;
    const int m0 = (blockIdx.x * 8 + wave) * 16;

    // --- A fragments: row = m0 + lrow, K runs [kb, kb+8) and [kb+16, kb+24)
    v16bf a[KPAD / 32];
    const __bf16* xr = X + (size_t)(m0 + lrow) * KPAD + half * 8;
    #pragma unroll
    for (int kt = 0; kt < KPAD / 32; ++kt) {
        bf16x8 lo = *(const bf16x8*)(xr + kt * 32);
        bf16x8 hi = *(const bf16x8*)(xr + kt * 32 + 16);
        #pragma unroll
        for (int e = 0; e < 8; ++e) { a[kt][e] = lo[e]; a[kt][e + 8] = hi[e]; }
    }

    #pragma unroll
    for (int nt = 0; nt < NOUT / 16; ++nt) {
        const int n = nt * 16 + lrow;
        const float bv = bias[n];
        v8f c;
        #pragma unroll
        for (int r = 0; r < 8; ++r) c[r] = bv;

        // --- B fragments: column n of W, K runs [kb, kb+8) and [kb+16, kb+24)
        const __bf16* wr = W + (size_t)n * KPAD + half * 8;
        #pragma unroll
        for (int kt = 0; kt < KPAD / 32; ++kt) {
            bf16x8 blo = *(const bf16x8*)(wr + kt * 32);
            bf16x8 bhi = *(const bf16x8*)(wr + kt * 32 + 16);
            v16bf bfrag;
            #pragma unroll
            for (int e = 0; e < 8; ++e) { bfrag[e] = blo[e]; bfrag[e + 8] = bhi[e]; }
            c = __builtin_amdgcn_wmma_f32_16x16x32_bf16(
                    false, a[kt], false, bfrag, (short)0, c, false, false);
        }

        // --- epilogue: store bf16 h, accumulate per-channel stats
        float psum = 0.0f, psq = 0.0f;
        #pragma unroll
        for (int r = 0; r < 8; ++r) {
            float v = c[r];
            int m = m0 + half * 8 + r;
            H[(size_t)m * NOUT + n] = (__bf16)v;
            psum += v; psq += v * v;
        }
        atomicAdd(&s_sum[n], psum);
        atomicAdd(&s_sq[n], psq);
    }
    __syncthreads();
    if (tid < NOUT) {
        atomicAdd(&stats[tid], s_sum[tid]);
        atomicAdd(&stats[NOUT + tid], s_sq[tid]);
    }
}

// ---------------------------------------------------------------------------
// 6) Fold BN into per-channel scale/shift:  y = h*scale + shift, relu.
// ---------------------------------------------------------------------------
__global__ void finalize_stats_kernel(const float* __restrict__ stats,
                                      const float* __restrict__ g,
                                      const float* __restrict__ be,
                                      float* __restrict__ scale,
                                      float* __restrict__ shift,
                                      int nout)
{
    const int o = threadIdx.x;
    if (o < nout) {
        const float invM = 1.0f / (float)MROWS;
        float mu  = stats[o] * invM;
        float var = stats[nout + o] * invM - mu * mu;
        float s   = g[o] * rsqrtf(var + BN_EPS);
        scale[o]  = s;
        shift[o]  = be[o] - mu * s;
    }
}

// ---------------------------------------------------------------------------
// 7) BN + ReLU applied elementwise, producing bf16 input for next layer.
// ---------------------------------------------------------------------------
template <int NOUT>
__global__ __launch_bounds__(256) void bn_relu_kernel(const __bf16* __restrict__ H,
                                                      const float* __restrict__ scale,
                                                      const float* __restrict__ shift,
                                                      __bf16* __restrict__ X)
{
    const size_t i = (size_t)blockIdx.x * 256 + threadIdx.x;
    if (i >= (size_t)MROWS * NOUT) return;
    const int o = (int)(i % NOUT);
    float v = (float)H[i] * scale[o] + shift[o];
    X[i] = (__bf16)fmaxf(v, 0.0f);
}

// ---------------------------------------------------------------------------
// 8) Final BN + ReLU + max over the NSAMPLE group -> new_points (f32).
// ---------------------------------------------------------------------------
__global__ __launch_bounds__(256) void final_max_kernel(const __bf16* __restrict__ H2,
                                                        const float* __restrict__ scale,
                                                        const float* __restrict__ shift,
                                                        float* __restrict__ out)
{
    const int t = blockIdx.x * 256 + threadIdx.x;   // BATCH*NPOINT*128
    if (t >= BATCH * NPOINT * 128) return;
    const int o = t & 127;
    const int bs = t >> 7;
    const __bf16* p = H2 + (size_t)bs * NSAMPLE * 128 + o;
    const float sc = scale[o], sh = shift[o];
    float m = 0.0f;                                  // relu output is >= 0
    #pragma unroll 8
    for (int k = 0; k < NSAMPLE; ++k) {
        float v = fmaxf((float)p[(size_t)k * 128] * sc + sh, 0.0f);
        m = fmaxf(m, v);
    }
    out[t] = m;
}

// ---------------------------------------------------------------------------
// Host side
// ---------------------------------------------------------------------------
extern "C" void kernel_launch(void* const* d_in, const int* in_sizes, int n_in,
                              void* d_out, int out_size, void* d_ws, size_t ws_size,
                              hipStream_t stream)
{
    const float* xyz    = (const float*)d_in[0];
    const float* points = (const float*)d_in[1];
    const float* w0 = (const float*)d_in[2];  const float* b0 = (const float*)d_in[3];
    const float* g0 = (const float*)d_in[4];  const float* be0 = (const float*)d_in[5];
    const float* w1 = (const float*)d_in[6];  const float* b1 = (const float*)d_in[7];
    const float* g1 = (const float*)d_in[8];  const float* be1 = (const float*)d_in[9];
    const float* w2 = (const float*)d_in[10]; const float* b2 = (const float*)d_in[11];
    const float* g2 = (const float*)d_in[12]; const float* be2 = (const float*)d_in[13];

    float* out        = (float*)d_out;
    float* new_xyz    = out;                               // 16*1024*3
    float* new_points = out + (size_t)BATCH * NPOINT * 3;  // 16*1024*128

    // ---- workspace carve-up (256B aligned) ----
    char* ws = (char*)d_ws;
    size_t off = 0;
    auto alloc = [&](size_t bytes) -> char* {
        char* p = ws + off;
        off = (off + bytes + 255) & ~(size_t)255;
        return p;
    };
    int*    fps_idx = (int*)alloc((size_t)BATCH * NPOINT * 4);
    int*    idx     = (int*)alloc((size_t)MROWS * 4);
    __bf16* W0p     = (__bf16*)alloc(64 * 96 * 2);
    __bf16* W1p     = (__bf16*)alloc(64 * 64 * 2);
    __bf16* W2p     = (__bf16*)alloc(128 * 64 * 2);
    float*  stats   = (float*)alloc(256 * 4);
    float*  scale   = (float*)alloc(128 * 4);
    float*  shift   = (float*)alloc(128 * 4);
    __bf16* X0      = (__bf16*)alloc((size_t)MROWS * 96 * 2);   // gp (padded)
    __bf16* H0      = (__bf16*)alloc((size_t)MROWS * 64 * 2);   // layer0 out
    __bf16* X1      = (__bf16*)alloc((size_t)MROWS * 64 * 2);   // relu(bn(h0))
    __bf16* H2      = (__bf16*)alloc((size_t)MROWS * 128 * 2);  // layer2 out
    __bf16* H1 = X0;   // X0 dead after gemm0 -> reuse for layer1 output
    __bf16* X2 = H0;   // H0 dead after bn_relu0 -> reuse for layer2 input
    (void)ws_size; (void)in_sizes; (void)n_in; (void)out_size; (void)fps_idx;

    // ---- geometry phase ----
    fps_kernel<<<BATCH, 256, 0, stream>>>(xyz, fps_idx, new_xyz);
    qbp_kernel<<<BATCH * 4, 256, 0, stream>>>(xyz, new_xyz, idx);
    build_gp_kernel<<<MROWS / 256, 256, 0, stream>>>(xyz, points, new_xyz, idx, X0);
    prep_weights_kernel<<<(64 * 96 + 255) / 256, 256, 0, stream>>>(w0, w1, w2, W0p, W1p, W2p);

    const int gemm_blocks = MROWS / 128;   // 8 waves * 16 rows per block

    // ---- layer 0: 96 -> 64 ----
    hipMemsetAsync(stats, 0, 256 * 4, stream);
    gemm_bn_stats_kernel<96, 64><<<gemm_blocks, 256, 0, stream>>>(X0, W0p, b0, H0, stats);
    finalize_stats_kernel<<<1, 128, 0, stream>>>(stats, g0, be0, scale, shift, 64);
    bn_relu_kernel<64><<<(int)(((size_t)MROWS * 64 + 255) / 256), 256, 0, stream>>>(H0, scale, shift, X1);

    // ---- layer 1: 64 -> 64 ----
    hipMemsetAsync(stats, 0, 256 * 4, stream);
    gemm_bn_stats_kernel<64, 64><<<gemm_blocks, 256, 0, stream>>>(X1, W1p, b1, H1, stats);
    finalize_stats_kernel<<<1, 128, 0, stream>>>(stats, g1, be1, scale, shift, 64);
    bn_relu_kernel<64><<<(int)(((size_t)MROWS * 64 + 255) / 256), 256, 0, stream>>>(H1, scale, shift, X2);

    // ---- layer 2: 64 -> 128, then max over NSAMPLE ----
    hipMemsetAsync(stats, 0, 256 * 4, stream);
    gemm_bn_stats_kernel<64, 128><<<gemm_blocks, 256, 0, stream>>>(X2, W2p, b2, H2, stats);
    finalize_stats_kernel<<<1, 128, 0, stream>>>(stats, g2, be2, scale, shift, 128);
    final_max_kernel<<<(BATCH * NPOINT * 128) / 256, 256, 0, stream>>>(H2, scale, shift, new_points);
}
